// HierarchicalRoPEAttention_28278064677170
// MI455X (gfx1250) — matrix operand order repaired and checked by reading, not compiled, not run on hardware
//
#include <hip/hip_runtime.h>
#include <hip/hip_bf16.h>

typedef __attribute__((ext_vector_type(16))) _Float16 v16h;
typedef __attribute__((ext_vector_type(8)))  _Float16 v8h;
typedef __attribute__((ext_vector_type(8)))  float    v8f;
typedef __attribute__((ext_vector_type(4)))  unsigned int u32x4;
typedef __attribute__((ext_vector_type(8)))  int       i32x8;
typedef __attribute__((ext_vector_type(4)))  int       i32x4;

// ---------------------------------------------------------------------------
// WMMA helpers (CDNA5 16x16x32 f16 -> f32)
// ---------------------------------------------------------------------------
__device__ __forceinline__ v8f wmma32(v16h a, v16h b, v8f c) {
  return __builtin_amdgcn_wmma_f32_16x16x32_f16(
      /*neg_a=*/false, a, /*neg_b=*/false, b,
      /*c_mod=*/(short)0, c, /*reuse_a=*/false, /*reuse_b=*/false);
}

// A-matrix fragment (16x32, f16), source is row-major [16][ldk] (K contiguous).
__device__ __forceinline__ v16h frag_a16(const _Float16* base, int lane, int ldk) {
  int m  = lane & 15;
  int k0 = (lane & 16) >> 1;          // 0 or 8
  const _Float16* p = base + m * ldk + k0;
  v8h lo = *(const v8h*)(p);
  v8h hi = *(const v8h*)(p + 16);
  v16h r;
#pragma unroll
  for (int i = 0; i < 8; ++i) { r[i] = lo[i]; r[i + 8] = hi[i]; }
  return r;
}

// B-matrix fragment (32x16, f16), source is N-major [16][ldk] (B^T rows).
__device__ __forceinline__ v16h frag_b16(const _Float16* base, int lane, int ldk) {
  int n  = lane & 15;
  int kb = (lane & 16) >> 1;          // 0 or 8
  const _Float16* p = base + n * ldk + kb;
  v8h lo = *(const v8h*)(p);
  v8h hi = *(const v8h*)(p + 16);
  v16h r;
#pragma unroll
  for (int i = 0; i < 8; ++i) { r[i] = lo[i]; r[i + 8] = hi[i]; }
  return r;
}

// ---------------------------------------------------------------------------
// DPP16 cross-lane reductions over a 16-lane row (EXEC all-ones under WMMA).
// ---------------------------------------------------------------------------
#define DPP_F32(x, ctrl)                                                      \
  __builtin_bit_cast(float, __builtin_amdgcn_mov_dpp(                         \
      __builtin_bit_cast(int, (x)), (ctrl), 0xF, 0xF, true))

__device__ __forceinline__ float row16_max(float x) {
  x = fmaxf(x, DPP_F32(x, 0xB1));    // quad_perm(1,0,3,2)
  x = fmaxf(x, DPP_F32(x, 0x4E));    // quad_perm(2,3,0,1)
  x = fmaxf(x, DPP_F32(x, 0x141));   // row_half_mirror
  x = fmaxf(x, DPP_F32(x, 0x140));   // row_mirror
  return x;
}
__device__ __forceinline__ float row16_sum(float x) {
  x += DPP_F32(x, 0xB1);
  x += DPP_F32(x, 0x4E);
  x += DPP_F32(x, 0x141);
  x += DPP_F32(x, 0x140);
  return x;
}

// ---------------------------------------------------------------------------
// Tensor Data Mover: 2D tile (global, row-major, f16) -> LDS (rows packed).
// D# per CDNA5 ISA 08_async_tensor 8.3/8.4:
//   group0: count=1 | lds_addr | global_addr | type=2
//   group1: data_size=1 (2B), tensor_dim0/1, tile_dim0/1, tensor_dim0_stride
// groups 2/3 zero (2D: tile_dim2=0 -> unused).  Tracked by TENSORcnt.
// ---------------------------------------------------------------------------
__device__ __forceinline__ void tdm_load_tile_f16(const _Float16* gptr,
                                                  unsigned lds_addr,
                                                  unsigned tensor_d0, unsigned tensor_d1,
                                                  unsigned tile_d0, unsigned tile_d1,
                                                  unsigned stride0) {
  unsigned long long ga = (unsigned long long)(uintptr_t)gptr;
  u32x4 g0;
  g0[0] = 1u;                                                   // count=1, user
  g0[1] = lds_addr;                                             // LDS byte addr
  g0[2] = (unsigned)ga;                                         // gaddr[31:0]
  g0[3] = ((unsigned)(ga >> 32) & 0x01FFFFFFu) | (2u << 30);    // gaddr[56:32]|type=2
  i32x8 g1;
  g1[0] = (int)(1u << 16);                                      // data_size=1 (2B)
  g1[1] = (int)((tensor_d0 & 0xFFFFu) << 16);                   // tensor_dim0 lo16
  g1[2] = (int)((tensor_d0 >> 16) | ((tensor_d1 & 0xFFFFu) << 16));
  g1[3] = (int)((tensor_d1 >> 16) | (tile_d0 << 16));           // tile_dim0
  g1[4] = (int)(tile_d1 & 0xFFFFu);                             // tile_dim1, tile_dim2=0
  g1[5] = (int)stride0;                                         // dim0 stride lo32
  g1[6] = 0;
  g1[7] = 0;
  i32x4 z4 = {0, 0, 0, 0};
#if __has_include(<hip/amd_detail/amd_gfx1250_TDM.h>)
  i32x8 z8 = {0, 0, 0, 0, 0, 0, 0, 0};
  __builtin_amdgcn_tensor_load_to_lds(g0, g1, z4, z4, z8, 0);
#else
  __builtin_amdgcn_tensor_load_to_lds(g0, g1, z4, z4, 0);
#endif
}

__device__ __forceinline__ unsigned lds_addr_of(const void* p) {
  // LDS aperture maps addr[31:0] to the LDS byte offset (ISA 10.2).
  return (unsigned)(uintptr_t)p;
}

// ---------------------------------------------------------------------------
// Elementwise conversion kernels
// ---------------------------------------------------------------------------
__global__ __launch_bounds__(256) void cvt_f32_to_f16(const float* __restrict__ in,
                                                      _Float16* __restrict__ out, int n) {
  int i = blockIdx.x * 256 + threadIdx.x;
  if (i < n) out[i] = (_Float16)in[i];
}

// out[n*K + k] = in[k*N + n]   (weights [K][N] -> N-major [N][K], f16)
__global__ __launch_bounds__(256) void transpose_cvt(const float* __restrict__ in,
                                                     _Float16* __restrict__ out,
                                                     int K, int N) {
  int o = blockIdx.x * 256 + threadIdx.x;
  if (o >= K * N) return;
  int n = o / K;
  int k = o - n * K;
  out[o] = (_Float16)in[k * N + n];
}

// Averaged multi-scale cos/sin tables: [T][32]
__global__ __launch_bounds__(256) void rope_tables(float* __restrict__ cosT,
                                                   float* __restrict__ sinT, int T) {
  int idx = blockIdx.x * 256 + threadIdx.x;
  if (idx >= T * 32) return;
  int t = idx >> 5, i = idx & 31;
  const float scales[3] = {1.0f, 4.0f, 16.0f};
  float c = 0.0f, s = 0.0f;
#pragma unroll
  for (int j = 0; j < 3; ++j) {
    float f = __expf(-(2.0f * (float)i / 64.0f) * __logf(10000.0f * scales[j]));
    float ang = (float)t * f;
    c += cosf(ang);
    s += sinf(ang);
  }
  cosT[idx] = c * (1.0f / 3.0f);
  sinT[idx] = s * (1.0f / 3.0f);
}

// Apply RoPE to q,k (f16 GEMM outputs [B*T][H*64]) and repack:
//   qh,kh : [B][H][T][64];  vt : [B][H][64][T]
__global__ __launch_bounds__(256) void rope_layout(const _Float16* __restrict__ qg,
                                                   const _Float16* __restrict__ kg,
                                                   const _Float16* __restrict__ vg,
                                                   const float* __restrict__ cosT,
                                                   const float* __restrict__ sinT,
                                                   _Float16* __restrict__ qh,
                                                   _Float16* __restrict__ kh,
                                                   _Float16* __restrict__ vt,
                                                   int B, int H, int T) {
  int idx = blockIdx.x * 256 + threadIdx.x;          // over B*H*T*32
  if (idx >= B * H * T * 32) return;
  int i = idx & 31;
  int t = (idx >> 5) % T;
  int h = (idx / (32 * T)) % H;
  int b = idx / (32 * T * H);

  size_t src = ((size_t)(b * T + t)) * 1024 + (size_t)h * 64;
  float c = cosT[t * 32 + i];
  float s = sinT[t * 32 + i];

  size_t dst = (((size_t)(b * H + h)) * T + t) * 64;

  float xe = (float)qg[src + 2 * i], xo = (float)qg[src + 2 * i + 1];
  qh[dst + 2 * i]     = (_Float16)(xe * c - xo * s);
  qh[dst + 2 * i + 1] = (_Float16)(xe * s + xo * c);

  xe = (float)kg[src + 2 * i]; xo = (float)kg[src + 2 * i + 1];
  kh[dst + 2 * i]     = (_Float16)(xe * c - xo * s);
  kh[dst + 2 * i + 1] = (_Float16)(xe * s + xo * c);

  size_t vb = ((size_t)(b * H + h)) * 64 * T;
  vt[vb + (size_t)(2 * i) * T + t]     = vg[src + 2 * i];
  vt[vb + (size_t)(2 * i + 1) * T + t] = vg[src + 2 * i + 1];
}

// ---------------------------------------------------------------------------
// Tiled WMMA GEMM with TDM double buffering:
//   C[M][N] = A[M][K] * B[K][N], A f16 [M][K], BT f16 [N][K].
// Block 256 thr (8 waves), tile 128x64, BK=32; wave tile 32x32 (2x2 WMMA).
// Wave 0 issues tensor_load_to_lds for tile i+1 while all waves compute tile i;
// pipeline depth enforced with s_wait_tensorcnt.
// ---------------------------------------------------------------------------
template <typename OutT>
__global__ __launch_bounds__(256) void gemm_wmma(const _Float16* __restrict__ A,
                                                 const _Float16* __restrict__ BT,
                                                 OutT* __restrict__ C,
                                                 int M, int N, int K) {
  __shared__ _Float16 As[2][128 * 32];
  __shared__ _Float16 Bs[2][64 * 32];

  int tid  = threadIdx.x;
  int lane = tid & 31;
  int wave = tid >> 5;
  int wm   = wave >> 1;        // 0..3
  int wn   = wave & 1;         // 0..1
  int bm0  = blockIdx.y * 128;
  int bn0  = blockIdx.x * 64;
  bool leader = (tid < 32);    // wave 0 drives the TDM

  v8f acc[2][2];
#pragma unroll
  for (int i = 0; i < 2; ++i)
#pragma unroll
    for (int j = 0; j < 2; ++j) acc[i][j] = v8f{0.f, 0.f, 0.f, 0.f, 0.f, 0.f, 0.f, 0.f};

  int nk = K / 32;
  if (leader) {
    tdm_load_tile_f16(A  + (size_t)bm0 * K, lds_addr_of(&As[0][0]),
                      (unsigned)K, (unsigned)M, 32u, 128u, (unsigned)K);
    tdm_load_tile_f16(BT + (size_t)bn0 * K, lds_addr_of(&Bs[0][0]),
                      (unsigned)K, (unsigned)N, 32u, 64u, (unsigned)K);
  }

  for (int i = 0; i < nk; ++i) {
    int cur = i & 1;
    if (leader) {
      if (i + 1 < nk) {
        int k1 = (i + 1) * 32;
        tdm_load_tile_f16(A  + (size_t)bm0 * K + k1, lds_addr_of(&As[cur ^ 1][0]),
                          (unsigned)K, (unsigned)M, 32u, 128u, (unsigned)K);
        tdm_load_tile_f16(BT + (size_t)bn0 * K + k1, lds_addr_of(&Bs[cur ^ 1][0]),
                          (unsigned)K, (unsigned)N, 32u, 64u, (unsigned)K);
        __builtin_amdgcn_s_wait_tensorcnt(2);   // tile i complete, i+1 in flight
      } else {
        __builtin_amdgcn_s_wait_tensorcnt(0);   // last tile complete
      }
    }
    __syncthreads();                            // LDS tile i visible to all waves

    v16h a0 = frag_a16(&As[cur][0] + (wm * 32 + 0)  * 32, lane, 32);
    v16h a1 = frag_a16(&As[cur][0] + (wm * 32 + 16) * 32, lane, 32);
    v16h b0 = frag_b16(&Bs[cur][0] + (wn * 32 + 0)  * 32, lane, 32);
    v16h b1 = frag_b16(&Bs[cur][0] + (wn * 32 + 16) * 32, lane, 32);

    acc[0][0] = wmma32(a0, b0, acc[0][0]);
    acc[0][1] = wmma32(a0, b1, acc[0][1]);
    acc[1][0] = wmma32(a1, b0, acc[1][0]);
    acc[1][1] = wmma32(a1, b1, acc[1][1]);

    __syncthreads();                            // reads of buf[cur] done
  }

  int mlo = (lane & 16) >> 1;   // 0 or 8
  int nlo = lane & 15;
#pragma unroll
  for (int i = 0; i < 2; ++i)
#pragma unroll
    for (int j = 0; j < 2; ++j)
#pragma unroll
      for (int r = 0; r < 8; ++r) {
        int row = bm0 + wm * 32 + i * 16 + mlo + r;
        int col = bn0 + wn * 32 + j * 16 + nlo;
        C[(size_t)row * N + col] = (OutT)acc[i][j][r];
      }
}

// ---------------------------------------------------------------------------
// Flash attention: one wave (32 thr) per 16-query block per (b,h).
// 64 keys per iteration: 8 S-WMMAs + online softmax (DPP reductions) + 8 O-WMMAs.
// qh,kh: [B][H][T][64] f16; vt: [B][H][64][T] f16; ao: [B*T][H*64] f16.
// ---------------------------------------------------------------------------
__global__ __launch_bounds__(32) void attn_flash(const _Float16* __restrict__ qh,
                                                 const _Float16* __restrict__ kh,
                                                 const _Float16* __restrict__ vt,
                                                 _Float16* __restrict__ ao,
                                                 int B, int H, int T) {
  __shared__ _Float16 Pt[16 * 64];

  int lane = threadIdx.x;
  int q0   = blockIdx.x * 16;
  int bh   = blockIdx.y;

  const _Float16* Q  = qh + (((size_t)bh * T) + q0) * 64;
  const _Float16* Kp = kh + (size_t)bh * T * 64;
  const _Float16* Vt = vt + (size_t)bh * 64 * T;

  v16h aq0 = frag_a16(Q,      lane, 64);   // K-dim 0..31
  v16h aq1 = frag_a16(Q + 32, lane, 64);   // K-dim 32..63

  v8f acc[4];
#pragma unroll
  for (int d = 0; d < 4; ++d) acc[d] = v8f{0.f, 0.f, 0.f, 0.f, 0.f, 0.f, 0.f, 0.f};
  float mrow[8], lrow[8];
#pragma unroll
  for (int r = 0; r < 8; ++r) { mrow[r] = -1.0e30f; lrow[r] = 0.0f; }

  const float scale = 0.125f;   // 1/sqrt(64)
  int nlo = lane & 15;
  int hi8 = (lane & 16) >> 1;   // 0 or 8

  for (int t0 = 0; t0 < T; t0 += 64) {
    // ---- S = Q K^T for 64 keys (4 n-tiles of 16) ----
    v8f s[4];
#pragma unroll
    for (int j = 0; j < 4; ++j) {
      const _Float16* Kb = Kp + (size_t)(t0 + j * 16) * 64;
      v8f sj = v8f{0.f, 0.f, 0.f, 0.f, 0.f, 0.f, 0.f, 0.f};
      sj = wmma32(aq0, frag_b16(Kb,      lane, 64), sj);
      sj = wmma32(aq1, frag_b16(Kb + 32, lane, 64), sj);
      s[j] = sj;
    }

    // ---- online softmax over the 64-key slab ----
    float corr[8];
#pragma unroll
    for (int r = 0; r < 8; ++r) {
      float e0 = s[0][r] * scale;
      float e1 = s[1][r] * scale;
      float e2 = s[2][r] * scale;
      float e3 = s[3][r] * scale;
      float mx = fmaxf(fmaxf(e0, e1), fmaxf(e2, e3));
      mx = row16_max(mx);                       // row max over 64 keys
      float nm = fmaxf(mrow[r], mx);
      float p0 = __expf(e0 - nm);
      float p1 = __expf(e1 - nm);
      float p2 = __expf(e2 - nm);
      float p3 = __expf(e3 - nm);
      float cr = __expf(mrow[r] - nm);
      float rs = row16_sum((p0 + p1) + (p2 + p3));
      lrow[r] = lrow[r] * cr + rs;
      mrow[r] = nm;
      corr[r] = cr;
      _Float16* pr = Pt + (hi8 + r) * 64 + nlo;
      pr[0]  = (_Float16)p0;
      pr[16] = (_Float16)p1;
      pr[32] = (_Float16)p2;
      pr[48] = (_Float16)p3;
    }
#pragma unroll
    for (int d = 0; d < 4; ++d)
#pragma unroll
      for (int r = 0; r < 8; ++r) acc[d][r] *= corr[r];

    __syncthreads();                     // P (C-layout) -> LDS -> A-layout
    v16h ap0 = frag_a16(Pt,      lane, 64);   // keys t0..t0+31
    v16h ap1 = frag_a16(Pt + 32, lane, 64);   // keys t0+32..t0+63
#pragma unroll
    for (int d = 0; d < 4; ++d) {
      const _Float16* Vb = Vt + (size_t)(d * 16) * T + t0;
      acc[d] = wmma32(ap0, frag_b16(Vb,      lane, T), acc[d]);
      acc[d] = wmma32(ap1, frag_b16(Vb + 32, lane, T), acc[d]);
    }
    __syncthreads();
  }

  int b = bh / H, h = bh % H;
#pragma unroll
  for (int r = 0; r < 8; ++r) {
    int row = q0 + hi8 + r;
    float inv = 1.0f / lrow[r];
#pragma unroll
    for (int d = 0; d < 4; ++d) {
      ao[(size_t)(b * T + row) * 1024 + h * 64 + d * 16 + nlo] =
          (_Float16)(acc[d][r] * inv);
    }
  }
}

// ---------------------------------------------------------------------------
// Launch
// ---------------------------------------------------------------------------
extern "C" void kernel_launch(void* const* d_in, const int* in_sizes, int n_in,
                              void* d_out, int out_size, void* d_ws, size_t ws_size,
                              hipStream_t stream) {
  const int B = 2, T = 2048, D = 1024, H = 16;
  const int M = B * T;                    // 4096

  const float* x  = (const float*)d_in[0];
  const float* Wq = (const float*)d_in[1];
  const float* Wk = (const float*)d_in[2];
  const float* Wv = (const float*)d_in[3];
  const float* Wo = (const float*)d_in[4];
  float* out = (float*)d_out;

  char* ws = (char*)d_ws;
  const size_t MD2 = (size_t)M * D * 2;   // f16 [4096][1024] = 8 MB
  const size_t DD2 = (size_t)D * D * 2;   // f16 [1024][1024] = 2 MB

  _Float16* xh  = (_Float16*)ws;            ws += MD2;
  _Float16* wqT = (_Float16*)ws;            ws += DD2;
  _Float16* wkT = (_Float16*)ws;            ws += DD2;
  _Float16* wvT = (_Float16*)ws;            ws += DD2;
  _Float16* woT = (_Float16*)ws;            ws += DD2;
  _Float16* qg  = (_Float16*)ws;            ws += MD2;
  _Float16* kg  = (_Float16*)ws;            ws += MD2;
  _Float16* vg  = (_Float16*)ws;            ws += MD2;
  _Float16* qhb = (_Float16*)ws;            ws += MD2;
  _Float16* khb = (_Float16*)ws;            ws += MD2;
  _Float16* vtb = (_Float16*)ws;            ws += MD2;
  float* cosT   = (float*)ws;               ws += (size_t)T * 32 * 4;
  float* sinT   = (float*)ws;               ws += (size_t)T * 32 * 4;
  _Float16* aoh = xh;                       // reuse: x no longer needed post-QKV

  // 1) conversions
  cvt_f32_to_f16<<<(M * D + 255) / 256, 256, 0, stream>>>(x, xh, M * D);
  transpose_cvt<<<(D * D + 255) / 256, 256, 0, stream>>>(Wq, wqT, D, D);
  transpose_cvt<<<(D * D + 255) / 256, 256, 0, stream>>>(Wk, wkT, D, D);
  transpose_cvt<<<(D * D + 255) / 256, 256, 0, stream>>>(Wv, wvT, D, D);
  transpose_cvt<<<(D * D + 255) / 256, 256, 0, stream>>>(Wo, woT, D, D);
  rope_tables<<<(T * 32 + 255) / 256, 256, 0, stream>>>(cosT, sinT, T);

  // 2) QKV projections (f16 WMMA, f32 accumulate -> f16 out)
  dim3 ggrid(D / 64, M / 128);
  gemm_wmma<_Float16><<<ggrid, 256, 0, stream>>>(xh, wqT, qg, M, D, D);
  gemm_wmma<_Float16><<<ggrid, 256, 0, stream>>>(xh, wkT, kg, M, D, D);
  gemm_wmma<_Float16><<<ggrid, 256, 0, stream>>>(xh, wvT, vg, M, D, D);

  // 3) RoPE + head-major repack (+ V transpose)
  rope_layout<<<(B * H * T * 32 + 255) / 256, 256, 0, stream>>>(
      qg, kg, vg, cosT, sinT, qhb, khb, vtb, B, H, T);

  // 4) flash attention (WMMA, one wave per 16-query block)
  attn_flash<<<dim3(T / 16, B * H), 32, 0, stream>>>(qhb, khb, vtb, aoh, B, H, T);

  // 5) output projection -> f32 d_out
  gemm_wmma<float><<<ggrid, 256, 0, stream>>>(aoh, woT, out, M, D, D);
}
